// PhonemeRVQTokenizer_16587163697608
// MI455X (gfx1250) — matrix operand order repaired and checked by reading, not compiled, not run on hardware
//
#include <hip/hip_runtime.h>

typedef __attribute__((ext_vector_type(16))) _Float16 v16h;
typedef __attribute__((ext_vector_type(8)))  float    v8f;

#define B_    2048
#define F_    64
#define H_    128
#define L_    256
#define D_    256
#define KCB_  1024
#define NQ_   8
#define MF_   (B_*F_)      // 131072 rows
#define RECON_ (B_*F_*14)

// ---------------- WMMA fragment helpers (contiguous k order, same in A and B) ----------------
// A (16x32 f16): lane L holds row L%16, K-chunk of 16 starting at (L/16)*16
// B (32x16 f16): lane L holds col L%16, K-chunk of 16 starting at (L/16)*16. B is stored K-major
// (column stride = full K), so the fragment is 16 contiguous halves -> one pair of b128 loads.

__device__ __forceinline__ v16h frag_a_ptr(const _Float16* p) {   // p: 16B aligned
  v16h r;
  *(uint4*)&r       = *(const uint4*)p;
  *((uint4*)&r + 1) = *(const uint4*)(p + 8);
  return r;
}

__device__ __forceinline__ v16h frag_b_ptr(const _Float16* p) {   // p: 16B aligned
  v16h r;
  *(uint4*)&r       = *(const uint4*)p;
  *((uint4*)&r + 1) = *(const uint4*)(p + 8);
  return r;
}

__device__ __forceinline__ v8f wmma16(v16h a, v16h b, v8f c) {
  return __builtin_amdgcn_wmma_f32_16x16x32_f16(false, a, false, b, (short)0, c, false, false);
}

__device__ __forceinline__ float gelu_f(float x) {
  return 0.5f * x * (1.0f + erff(x * 0.70710678118654752f));
}

// ---------------- generic WMMA GEMM: out = act(alpha*A@B + bias + bias2 + residual) ----------------
// A: row-major (M x K), Bt: K-major (N columns, each K contiguous halves, column stride ldb >= K).
// block = 4 waves; block tile 16x64. M%16==0, N%64==0, K%32==0.

__global__ __launch_bounds__(128) void gemm_kernel(
    const _Float16* __restrict__ A, int lda,
    const _Float16* __restrict__ Bt, int ldb,
    const float* __restrict__ bias, const float* __restrict__ bias2,
    const float* __restrict__ residual, int ldres,
    float alpha, int act,
    float* __restrict__ out32, _Float16* __restrict__ out16, int ldc, int K)
{
  int wave = threadIdx.x >> 5, lane = threadIdx.x & 31;
  int row0 = blockIdx.x * 16;
  int col0 = blockIdx.y * 64 + wave * 16;
  const _Float16* pA = A  + (size_t)(row0 + (lane & 15)) * lda + ((lane >> 4) << 4);
  const _Float16* pB = Bt + (size_t)(col0 + (lane & 15)) * ldb + ((lane >> 4) << 4);
  v8f acc = {};
  for (int k0 = 0; k0 < K; k0 += 32, pA += 32, pB += 32)
    acc = wmma16(frag_a_ptr(pA), frag_b_ptr(pB), acc);
  int col = col0 + (lane & 15);
  int rb  = row0 + ((lane >> 4) << 3);
  float badd = (bias ? bias[col] : 0.f) + (bias2 ? bias2[col] : 0.f);
#pragma unroll
  for (int r = 0; r < 8; ++r) {
    int row = rb + r;
    float v = alpha * acc[r] + badd;
    if (residual) v += residual[(size_t)row * ldres + col];
    if (act == 1) v = gelu_f(v);
    else if (act == 2) v = fmaxf(v, 0.f);
    if (out32) out32[(size_t)row * ldc + col] = v;
    if (out16) out16[(size_t)row * ldc + col] = (_Float16)v;
  }
}

// ---------------- elementwise / LN / small kernels ----------------

__global__ void cvt_kernel(const float* __restrict__ s, _Float16* __restrict__ d, size_t n) {
  size_t i = (size_t)blockIdx.x * blockDim.x + threadIdx.x;
  if (i < n) d[i] = (_Float16)s[i];
}

// transpose-convert: src row-major (K x N) f32 -> dst K-major f16: dst[n*K+k] = src[k*N+n]
__global__ void cvtT_kernel(const float* __restrict__ s, _Float16* __restrict__ d, int K, int N) {
  int i = blockIdx.x * blockDim.x + threadIdx.x;
  if (i >= K * N) return;
  int k = i / N, n = i - k * N;
  d[(size_t)n * K + k] = (_Float16)s[i];
}

// y = LN(x [+ t]); write optional f32 and f16. blockDim == N (128 or 256), dyn-shared N floats.
__global__ void addln_kernel(const float* __restrict__ xin, const float* __restrict__ t,
                             const float* __restrict__ g, const float* __restrict__ bta,
                             _Float16* __restrict__ o16, float* __restrict__ o32)
{
  extern __shared__ float red[];
  int N = blockDim.x, row = blockIdx.x, c = threadIdx.x;
  size_t o = (size_t)row * N + c;
  float v = xin[o] + (t ? t[o] : 0.f);
  red[c] = v; __syncthreads();
  for (int s = N >> 1; s > 0; s >>= 1) { if (c < s) red[c] += red[c + s]; __syncthreads(); }
  float mean = red[0] / (float)N; __syncthreads();
  float d0 = v - mean;
  red[c] = d0 * d0; __syncthreads();
  for (int s = N >> 1; s > 0; s >>= 1) { if (c < s) red[c] += red[c + s]; __syncthreads(); }
  float var = red[0] / (float)N;
  float y = d0 * rsqrtf(var + 1e-5f) * g[c] + bta[c];
  if (o32) o32[o] = y;
  if (o16) o16[o] = (_Float16)y;
}

__global__ void phemb_kernel(const float* __restrict__ emb, const int* __restrict__ ids,
                             float* __restrict__ o32, _Float16* __restrict__ o16)
{
  int b = blockIdx.x, h = threadIdx.x;
  float v = emb[ids[b] * H_ + h];
  o32[(size_t)b * H_ + h] = v;
  o16[(size_t)b * H_ + h] = (_Float16)v;
}

__global__ __launch_bounds__(128) void enc_in_kernel(
    const float* __restrict__ frames, const float* __restrict__ Wp, const float* __restrict__ bp,
    const float* __restrict__ phemb, float* __restrict__ x32)
{
  int row = blockIdx.x, b = row >> 6, n = threadIdx.x;
  float s = bp[n] + phemb[(size_t)b * H_ + n];
  const float* fr = frames + (size_t)row * 14;
#pragma unroll
  for (int k = 0; k < 14; ++k) s += fr[k] * Wp[k * H_ + n];
  x32[(size_t)row * H_ + n] = s;
}

__global__ __launch_bounds__(128) void pool_kernel(
    const float* __restrict__ x32, const float* __restrict__ query, const int* __restrict__ lengths,
    int is_end, float* __restrict__ p32, _Float16* __restrict__ p16)
{
  __shared__ float sc[F_];
  __shared__ float mx, sm;
  int b = blockIdx.x, t = threadIdx.x;
  int len = lengths[b];
  int mid = len / 2; if (mid < 1) mid = 1;
  if (t < F_) {
    const float* xr = x32 + ((size_t)b * F_ + t) * H_;
    float s = 0.f;
    for (int k = 0; k < H_; ++k) s += xr[k] * query[k];
    s *= 0.08838834764831845f;  // 128^-0.5
    bool valid = t < len;
    bool msk_s = (t < mid) && valid;
    bool m = is_end ? (len == 1 ? msk_s : ((t >= mid) && valid)) : msk_s;
    sc[t] = m ? s : -1e9f;
  }
  __syncthreads();
  if (t == 0) {
    float m0 = -1e30f;
    for (int i = 0; i < F_; ++i) m0 = fmaxf(m0, sc[i]);
    float s0 = 0.f;
    for (int i = 0; i < F_; ++i) s0 += __expf(sc[i] - m0);
    mx = m0; sm = s0;
  }
  __syncthreads();
  if (t < F_) sc[t] = __expf(sc[t] - mx) / sm;
  __syncthreads();
  float acc = 0.f;
  for (int f = 0; f < F_; ++f) acc += sc[f] * x32[((size_t)b * F_ + f) * H_ + t];
  p32[(size_t)b * H_ + t] = acc;
  p16[(size_t)b * H_ + t] = (_Float16)acc;
}

__global__ void cnorm_kernel(const float* __restrict__ cb, float* __restrict__ cn, int total) {
  int i = blockIdx.x * blockDim.x + threadIdx.x;
  if (i >= total) return;
  const float* p = cb + (size_t)i * L_;
  float s = 0.f;
  for (int k = 0; k < L_; ++k) s += p[k] * p[k];
  cn[i] = s;
}

__global__ __launch_bounds__(256) void argmin_kernel(const float* __restrict__ dist,
                                                     int* __restrict__ idxb,
                                                     float* __restrict__ idx_out, int q)
{
  __shared__ float bv[256]; __shared__ int bix[256];
  int b = blockIdx.x, t = threadIdx.x;
  const float* d = dist + (size_t)b * KCB_;
  float best = 3.4e38f; int bi = 0;
  for (int k = t; k < KCB_; k += 256) { float v = d[k]; if (v < best) { best = v; bi = k; } }
  bv[t] = best; bix[t] = bi; __syncthreads();
  for (int s = 128; s > 0; s >>= 1) {
    if (t < s) {
      if (bv[t + s] < bv[t] || (bv[t + s] == bv[t] && bix[t + s] < bix[t])) { bv[t] = bv[t + s]; bix[t] = bix[t + s]; }
    }
    __syncthreads();
  }
  if (t == 0) { idxb[b] = bix[0]; idx_out[(size_t)b * NQ_ + q] = (float)bix[0]; }
}

__global__ __launch_bounds__(256) void rvq_update_kernel(
    const float* __restrict__ cb, const int* __restrict__ idxb,
    float* __restrict__ quant, float* __restrict__ r32, _Float16* __restrict__ r16,
    float* __restrict__ commit_acc)
{
  __shared__ float red[256];
  int b = blockIdx.x, t = threadIdx.x;
  int k = idxb[b];
  float c = cb[(size_t)k * L_ + t];
  size_t o = (size_t)b * L_ + t;
  float r = r32[o];
  float diff = c - r;
  quant[o] += c;
  float rn = r - c;
  r32[o] = rn;
  r16[o] = (_Float16)rn;
  red[t] = diff * diff; __syncthreads();
  for (int s = 128; s > 0; s >>= 1) { if (t < s) red[t] += red[t + s]; __syncthreads(); }
  if (t == 0) atomicAdd(commit_acc, red[0]);
}

__global__ __launch_bounds__(128) void sinpos_kernel(const int* __restrict__ lengths,
                                                     _Float16* __restrict__ sp16)
{
  int row = blockIdx.x, b = row >> 6, f = row & 63;
  int len = lengths[b];
  int den = len - 1; if (den < 1) den = 1;
  float alpha = (len == 1) ? 0.5f : (float)f / (float)den;
  int i = threadIdx.x;  // 0..127
  float dv = __expf((float)(2 * i) * (-9.210340371976184f / 256.f));
  float ang = alpha * 100.f * dv;
  size_t o = (size_t)row * D_ + 2 * i;
  sp16[o]     = (_Float16)__sinf(ang);
  sp16[o + 1] = (_Float16)__cosf(ang);
}

// ---------------- fused self-attention, one (batch, head) per block ----------------
// Weights K-major (column stride 256). V stored transposed in LDS so the P@V B-fragment
// is contiguous (ds_load_b128), as is the Q@K^T fragment (K row-major in LDS).

__global__ __launch_bounds__(128) void sa_kernel(
    const _Float16* __restrict__ h16,
    const _Float16* __restrict__ WqT, const float* __restrict__ bq,
    const _Float16* __restrict__ WkT, const float* __restrict__ bk,
    const _Float16* __restrict__ WvT, const float* __restrict__ bv,
    _Float16* __restrict__ o16)
{
  __shared__ __align__(16) _Float16 sh[F_ * D_];
  __shared__ __align__(16) _Float16 sq[F_ * 64], sk[F_ * 64], svT[F_ * 64];
  __shared__ float ss[F_ * F_];
  int b = blockIdx.x, head = blockIdx.y;
  size_t base = (size_t)b * F_ * D_;
  { const uint4* s = (const uint4*)(h16 + base); uint4* d = (uint4*)sh;
    for (int i = threadIdx.x; i < F_ * D_ / 8; i += 128) d[i] = s[i]; }
  __syncthreads();
  int wave = threadIdx.x >> 5, lane = threadIdx.x & 31;
  int row0 = wave * 16, colW = head * 64;
  int l15 = lane & 15, lhi = lane >> 4;
  for (int m = 0; m < 3; ++m) {
    const _Float16* WT = (m == 0) ? WqT : (m == 1 ? WkT : WvT);
    const float* bb = (m == 0) ? bq : (m == 1 ? bk : bv);
    for (int ct = 0; ct < 4; ++ct) {
      int col0 = ct * 16;
      const _Float16* pA = sh + (size_t)(row0 + l15) * D_ + (lhi << 4);
      const _Float16* pB = WT + (size_t)(colW + col0 + l15) * D_ + (lhi << 4);
      v8f acc = {};
      for (int k0 = 0; k0 < D_; k0 += 32, pA += 32, pB += 32)
        acc = wmma16(frag_a_ptr(pA), frag_b_ptr(pB), acc);
      int col = col0 + l15, rb = row0 + (lhi << 3);
      float ba = bb[colW + col];
      if (m == 2) {             // V transposed: svT[d*64 + frame]
#pragma unroll
        for (int r = 0; r < 8; ++r) svT[col * 64 + rb + r] = (_Float16)(acc[r] + ba);
      } else {
        _Float16* dst = (m == 0) ? sq : sk;
#pragma unroll
        for (int r = 0; r < 8; ++r) dst[(rb + r) * 64 + col] = (_Float16)(acc[r] + ba);
      }
    }
  }
  __syncthreads();
  for (int ct = 0; ct < 4; ++ct) {           // S = Q K^T * 0.125 ; B-frag = sk row-major (K-major)
    int col0 = ct * 16;
    const _Float16* pA = sq + (size_t)(row0 + l15) * 64 + (lhi << 4);
    const _Float16* pB = sk + (size_t)(col0 + l15) * 64 + (lhi << 4);
    v8f acc = {};
    for (int k0 = 0; k0 < 64; k0 += 32, pA += 32, pB += 32)
      acc = wmma16(frag_a_ptr(pA), frag_b_ptr(pB), acc);
    int col = col0 + l15, rb = row0 + (lhi << 3);
#pragma unroll
    for (int r = 0; r < 8; ++r) ss[(rb + r) * F_ + col] = acc[r] * 0.125f;
  }
  __syncthreads();
  if (threadIdx.x < F_) {                    // softmax rows, P -> sq (f16)
    float* sr = ss + threadIdx.x * F_;
    float m0 = -1e30f;
    for (int j = 0; j < F_; ++j) m0 = fmaxf(m0, sr[j]);
    float s0 = 0.f;
    for (int j = 0; j < F_; ++j) { float e = __expf(sr[j] - m0); sr[j] = e; s0 += e; }
    float inv = 1.f / s0;
    _Float16* pr = sq + threadIdx.x * 64;
    for (int j = 0; j < F_; ++j) pr[j] = (_Float16)(sr[j] * inv);
  }
  __syncthreads();
  for (int ct = 0; ct < 4; ++ct) {           // O = P V ; B-frag = svT (K-major)
    int col0 = ct * 16;
    const _Float16* pA = sq  + (size_t)(row0 + l15) * 64 + (lhi << 4);
    const _Float16* pB = svT + (size_t)(col0 + l15) * 64 + (lhi << 4);
    v8f acc = {};
    for (int k0 = 0; k0 < 64; k0 += 32, pA += 32, pB += 32)
      acc = wmma16(frag_a_ptr(pA), frag_b_ptr(pB), acc);
    int col = col0 + l15, rb = row0 + (lhi << 3);
#pragma unroll
    for (int r = 0; r < 8; ++r)
      o16[base + (size_t)(rb + r) * D_ + colW + col] = (_Float16)acc[r];
  }
}

// ---------------- fused cross-attention (Tk = 3), one (batch, head) per block ----------------

__global__ __launch_bounds__(128) void ca_kernel(
    const _Float16* __restrict__ h16,
    const _Float16* __restrict__ WqT, const float* __restrict__ bq,
    const _Float16* __restrict__ memK, const _Float16* __restrict__ memV,
    _Float16* __restrict__ o16)
{
  __shared__ __align__(16) _Float16 sh[F_ * D_];
  __shared__ float qf[F_ * 64];
  int b = blockIdx.x, head = blockIdx.y;
  size_t base = (size_t)b * F_ * D_;
  { const uint4* s = (const uint4*)(h16 + base); uint4* d = (uint4*)sh;
    for (int i = threadIdx.x; i < F_ * D_ / 8; i += 128) d[i] = s[i]; }
  __syncthreads();
  int wave = threadIdx.x >> 5, lane = threadIdx.x & 31;
  int row0 = wave * 16, colW = head * 64;
  int l15 = lane & 15, lhi = lane >> 4;
  for (int ct = 0; ct < 4; ++ct) {
    int col0 = ct * 16;
    const _Float16* pA = sh  + (size_t)(row0 + l15) * D_ + (lhi << 4);
    const _Float16* pB = WqT + (size_t)(colW + col0 + l15) * D_ + (lhi << 4);
    v8f acc = {};
    for (int k0 = 0; k0 < D_; k0 += 32, pA += 32, pB += 32)
      acc = wmma16(frag_a_ptr(pA), frag_b_ptr(pB), acc);
    int col = col0 + l15, rb = row0 + (lhi << 3);
#pragma unroll
    for (int r = 0; r < 8; ++r) qf[(rb + r) * 64 + col] = acc[r] + bq[colW + col];
  }
  __syncthreads();
  if (threadIdx.x < F_) {
    int r = threadIdx.x;
    const float* qr = qf + r * 64;
    float s[3];
    for (int j = 0; j < 3; ++j) {
      const _Float16* kr = memK + ((size_t)b * 3 + j) * D_ + colW;
      float a = 0.f;
      for (int d = 0; d < 64; ++d) a += qr[d] * (float)kr[d];
      s[j] = a * 0.125f;
    }
    float m0 = fmaxf(s[0], fmaxf(s[1], s[2]));
    float e0 = __expf(s[0] - m0), e1 = __expf(s[1] - m0), e2 = __expf(s[2] - m0);
    float inv = 1.f / (e0 + e1 + e2);
    e0 *= inv; e1 *= inv; e2 *= inv;
    const _Float16* v0 = memV + (size_t)b * 3 * D_ + colW;
    const _Float16* v1 = v0 + D_;
    const _Float16* v2 = v1 + D_;
    _Float16* orow = o16 + base + (size_t)r * D_ + colW;
    for (int n = 0; n < 64; ++n)
      orow[n] = (_Float16)(e0 * (float)v0[n] + e1 * (float)v1[n] + e2 * (float)v2[n]);
  }
}

// ---------------- fused FFN: x += relu(h@W1+b1)@W2+b2, hidden tile in LDS ----------------

__global__ __launch_bounds__(256) void ffn_kernel(
    const _Float16* __restrict__ h16,
    const _Float16* __restrict__ W1T, const float* __restrict__ b1,
    const _Float16* __restrict__ W2T, const float* __restrict__ b2,
    float* __restrict__ x)
{
  __shared__ __align__(16) _Float16 hid[32 * 1024];   // 64 KB
  int row0 = blockIdx.x * 32;
  int wave = threadIdx.x >> 5, lane = threadIdx.x & 31;
  int l15 = lane & 15, lhi = lane >> 4;
  for (int i = 0; i < 16; ++i) {                       // GEMM1: 32x1024x256
    int t = wave * 16 + i;
    int mr = (t & 1) * 16, nc = (t >> 1) * 16;
    const _Float16* pA = h16 + (size_t)(row0 + mr + l15) * D_ + (lhi << 4);
    const _Float16* pB = W1T + (size_t)(nc + l15) * D_ + (lhi << 4);
    v8f acc = {};
    for (int k0 = 0; k0 < D_; k0 += 32, pA += 32, pB += 32)
      acc = wmma16(frag_a_ptr(pA), frag_b_ptr(pB), acc);
    int col = nc + l15, rb = mr + (lhi << 3);
    float bb = b1[col];
#pragma unroll
    for (int r = 0; r < 8; ++r)
      hid[(rb + r) * 1024 + col] = (_Float16)fmaxf(acc[r] + bb, 0.f);
  }
  __syncthreads();
  for (int i = 0; i < 4; ++i) {                        // GEMM2: 32x256x1024
    int t = wave * 4 + i;
    int mr = (t & 1) * 16, nc = (t >> 1) * 16;
    const _Float16* pA = hid + (size_t)(mr + l15) * 1024 + (lhi << 4);
    const _Float16* pB = W2T + (size_t)(nc + l15) * 1024 + (lhi << 4);
    v8f acc = {};
    for (int k0 = 0; k0 < 1024; k0 += 32, pA += 32, pB += 32)
      acc = wmma16(frag_a_ptr(pA), frag_b_ptr(pB), acc);
    int col = nc + l15, rb = mr + (lhi << 3);
    float bb = b2[col];
#pragma unroll
    for (int r = 0; r < 8; ++r) {
      size_t o = (size_t)(row0 + rb + r) * D_ + col;
      x[o] += acc[r] + bb;
    }
  }
}

__global__ void outproj_kernel(const float* __restrict__ x, const float* __restrict__ W,
                               const float* __restrict__ bias, float* __restrict__ recon)
{
  int id = blockIdx.x * blockDim.x + threadIdx.x;
  if (id >= MF_ * 14) return;
  int row = id / 14, n = id % 14;
  const float* xr = x + (size_t)row * D_;
  float s = bias[n];
  for (int k = 0; k < D_; ++k) s += xr[k] * W[k * 14 + n];
  recon[id] = s;
}

__global__ void commit_kernel(const float* __restrict__ acc, float* __restrict__ out) {
  out[0] = acc[0] * (0.25f / ((float)B_ * (float)L_));
}

// =========================== host launcher ===========================

extern "C" void kernel_launch(void* const* d_in, const int* in_sizes, int n_in,
                              void* d_out, int out_size, void* d_ws, size_t ws_size,
                              hipStream_t stream)
{
  (void)in_sizes; (void)n_in; (void)out_size; (void)ws_size;
  auto F32 = [&](int i) { return (const float*)d_in[i]; };
  auto I32 = [&](int i) { return (const int*)d_in[i]; };

  // input index map (setup_inputs dict insertion order, tuples flattened)
  const int IN_FRAMES = 0, IN_PHIDS = 1, IN_LENGTHS = 2, IN_PHEMB = 3,
            IN_PH2DEC_W = 4, IN_PH2DEC_B = 5, ENC_S = 6, ENC_E = 21,
            IN_CB_S = 36, IN_CB_E = 37, IN_MEMPROJ_W = 38, IN_MEMPROJ_B = 39,
            IN_MEMTYPE = 40, IN_A1W = 41, IN_A1B = 42, IN_A2W = 43, IN_A2B = 44,
            LAY0 = 45, LAY1 = 71, IN_OUTW = 97, IN_OUTB = 98;
  // enc offsets: +0 projW +1 projB +2..7 norm0/1/2 (g,b) +8 l1W +9 l1B +10 l2W +11 l2B +12 query +13 outW +14 outB
  // layer offsets: saQ(w,b)0,1 saK2,3 saV4,5 saO6,7 caQ8,9 caK10,11 caV12,13 caO14,15 ffn1 16,17 ffn2 18,19 ln1 20,21 ln2 22,23 ln3 24,25

  char* w = (char*)d_ws;
  size_t off = 0;
  auto allocB = [&](size_t bytes) { void* p = w + off; off = (off + bytes + 255) & ~(size_t)255; return p; };
  auto allocF = [&](size_t n) { return (float*)allocB(n * 4); };
  auto allocH = [&](size_t n) { return (_Float16*)allocB(n * 2); };
  auto allocI = [&](size_t n) { return (int*)allocB(n * 4); };

  // f16 weight shadows, all stored K-major (pre-transposed)
  _Float16* w16_ph2dec = allocH(128 * 256);
  _Float16* sE_l1[2], * sE_l2[2], * sE_out[2];
  for (int e = 0; e < 2; ++e) { sE_l1[e] = allocH(128 * 128); sE_l2[e] = allocH(128 * 128); sE_out[e] = allocH(128 * 256); }
  _Float16* cb16s = allocH((size_t)NQ_ * KCB_ * L_);    // already K-major per entry
  _Float16* cb16e = allocH((size_t)NQ_ * KCB_ * L_);
  _Float16* w16_memproj = allocH(256 * 256);
  _Float16* w16_a1 = allocH(256 * 256);
  _Float16* w16_a2 = allocH(256 * 256);
  _Float16 *saQ16[2], *saK16[2], *saV16[2], *saO16[2], *caQ16[2], *caK16[2], *caV16[2], *caO16[2], *f1_16[2], *f2_16[2];
  for (int l = 0; l < 2; ++l) {
    saQ16[l] = allocH(65536); saK16[l] = allocH(65536); saV16[l] = allocH(65536); saO16[l] = allocH(65536);
    caQ16[l] = allocH(65536); caK16[l] = allocH(65536); caV16[l] = allocH(65536); caO16[l] = allocH(65536);
    f1_16[l] = allocH(256 * 1024); f2_16[l] = allocH(1024 * 256);
  }
  // activations / scratch
  float*     cnormS   = allocF(NQ_ * KCB_);
  float*     cnormE   = allocF(NQ_ * KCB_);
  float*     phemb32  = allocF((size_t)B_ * H_);
  _Float16*  phemb16  = allocH((size_t)B_ * H_);
  float*     encx32   = allocF((size_t)MF_ * H_);
  _Float16*  encx16   = allocH((size_t)MF_ * H_);
  float*     enct32   = allocF((size_t)MF_ * H_);
  float*     pooled32 = allocF((size_t)B_ * H_);
  _Float16*  pooled16 = allocH((size_t)B_ * H_);
  float*     r32      = allocF((size_t)B_ * L_);
  _Float16*  r16      = allocH((size_t)B_ * L_);
  float*     squant32 = allocF((size_t)B_ * L_);
  float*     equant32 = allocF((size_t)B_ * L_);
  _Float16*  squant16 = allocH((size_t)B_ * L_);
  _Float16*  equant16 = allocH((size_t)B_ * L_);
  float*     dist     = allocF((size_t)B_ * KCB_);
  int*       idxb     = allocI(B_);
  float*     commitA  = allocF(16);
  float*     mem32    = allocF((size_t)B_ * 3 * D_);
  _Float16*  mem16    = allocH((size_t)B_ * 3 * D_);
  _Float16*  memK16   = allocH((size_t)B_ * 3 * D_);
  _Float16*  memV16   = allocH((size_t)B_ * 3 * D_);
  float*     decx32   = allocF((size_t)MF_ * D_);
  _Float16*  h16A     = allocH((size_t)MF_ * D_);
  _Float16*  h16B     = allocH((size_t)MF_ * D_);

  float* outF    = (float*)d_out;
  float* reconO  = outF;
  float* sidxO   = outF + RECON_;
  float* eidxO   = sidxO + (size_t)B_ * NQ_;
  float* commitO = eidxO + (size_t)B_ * NQ_;

  auto cvt = [&](const float* s, _Float16* d, size_t n) {
    cvt_kernel<<<(unsigned)((n + 255) / 256), 256, 0, stream>>>(s, d, n);
  };
  auto cvtT = [&](const float* s, _Float16* d, int K, int N) {
    cvtT_kernel<<<(K * N + 255) / 256, 256, 0, stream>>>(s, d, K, N);
  };
  auto gemm = [&](const _Float16* A, int lda, const _Float16* Bt, int ldb,
                  const float* bias, const float* bias2, const float* res, int ldres,
                  float alpha, int act, float* o32, _Float16* o16, int ldc,
                  int M, int N, int K) {
    dim3 g(M / 16, N / 64);
    gemm_kernel<<<g, 128, 0, stream>>>(A, lda, Bt, ldb, bias, bias2, res, ldres,
                                       alpha, act, o32, o16, ldc, K);
  };

  // ---- weight conversions (transposed to K-major) ----
  cvtT(F32(IN_PH2DEC_W), w16_ph2dec, 128, 256);
  for (int e = 0; e < 2; ++e) {
    int base = e ? ENC_E : ENC_S;
    cvtT(F32(base + 8),  sE_l1[e], 128, 128);
    cvtT(F32(base + 10), sE_l2[e], 128, 128);
    cvtT(F32(base + 13), sE_out[e], 128, 256);
  }
  cvt(F32(IN_CB_S), cb16s, (size_t)NQ_ * KCB_ * L_);
  cvt(F32(IN_CB_E), cb16e, (size_t)NQ_ * KCB_ * L_);
  cvtT(F32(IN_MEMPROJ_W), w16_memproj, 256, 256);
  cvtT(F32(IN_A1W), w16_a1, 256, 256);
  cvtT(F32(IN_A2W), w16_a2, 256, 256);
  for (int l = 0; l < 2; ++l) {
    int L = l ? LAY1 : LAY0;
    cvtT(F32(L + 0),  saQ16[l], 256, 256); cvtT(F32(L + 2),  saK16[l], 256, 256);
    cvtT(F32(L + 4),  saV16[l], 256, 256); cvtT(F32(L + 6),  saO16[l], 256, 256);
    cvtT(F32(L + 8),  caQ16[l], 256, 256); cvtT(F32(L + 10), caK16[l], 256, 256);
    cvtT(F32(L + 12), caV16[l], 256, 256); cvtT(F32(L + 14), caO16[l], 256, 256);
    cvtT(F32(L + 16), f1_16[l], 256, 1024); cvtT(F32(L + 18), f2_16[l], 1024, 256);
  }
  cnorm_kernel<<<(NQ_ * KCB_ + 255) / 256, 256, 0, stream>>>(F32(IN_CB_S), cnormS, NQ_ * KCB_);
  cnorm_kernel<<<(NQ_ * KCB_ + 255) / 256, 256, 0, stream>>>(F32(IN_CB_E), cnormE, NQ_ * KCB_);

  hipMemsetAsync(commitA, 0, 4, stream);

  // ---- phoneme embedding + mem slot 2 (ph_dec + te2) ----
  phemb_kernel<<<B_, H_, 0, stream>>>(F32(IN_PHEMB), I32(IN_PHIDS), phemb32, phemb16);
  gemm(phemb16, 128, w16_ph2dec, 128, F32(IN_PH2DEC_B), F32(IN_MEMTYPE) + 512,
       nullptr, 0, 1.f, 0, mem32 + 512, nullptr, 768, B_, 256, 128);

  const float* frames = F32(IN_FRAMES);
  const int* lengths = I32(IN_LENGTHS);

  // ---- encoders + RVQ ----
  for (int e = 0; e < 2; ++e) {
    int base = e ? ENC_E : ENC_S;
    const _Float16* cb16c = e ? cb16e : cb16s;
    const float*    cbF   = e ? F32(IN_CB_E) : F32(IN_CB_S);
    const float*    cnc   = e ? cnormE : cnormS;
    float*          qz32  = e ? equant32 : squant32;
    _Float16*       qz16  = e ? equant16 : squant16;
    float*          idxO  = e ? eidxO : sidxO;

    enc_in_kernel<<<MF_, 128, 0, stream>>>(frames, F32(base + 0), F32(base + 1), phemb32, encx32);
    addln_kernel<<<MF_, 128, 512, stream>>>(encx32, nullptr, F32(base + 2), F32(base + 3), encx16, encx32);
    gemm(encx16, 128, sE_l1[e], 128, F32(base + 9), nullptr, nullptr, 0, 1.f, 1,
         enct32, nullptr, 128, MF_, 128, 128);
    addln_kernel<<<MF_, 128, 512, stream>>>(encx32, enct32, F32(base + 4), F32(base + 5), encx16, encx32);
    gemm(encx16, 128, sE_l2[e], 128, F32(base + 11), nullptr, nullptr, 0, 1.f, 1,
         enct32, nullptr, 128, MF_, 128, 128);
    addln_kernel<<<MF_, 128, 512, stream>>>(encx32, enct32, F32(base + 6), F32(base + 7), encx16, encx32);
    pool_kernel<<<B_, 128, 0, stream>>>(encx32, F32(base + 12), lengths, e, pooled32, pooled16);
    gemm(pooled16, 128, sE_out[e], 128, F32(base + 14), nullptr, nullptr, 0, 1.f, 0,
         r32, r16, 256, B_, 256, 128);

    hipMemsetAsync(qz32, 0, (size_t)B_ * L_ * 4, stream);
    for (int q = 0; q < NQ_; ++q) {
      gemm(r16, 256, cb16c + (size_t)q * KCB_ * L_, L_, cnc + (size_t)q * KCB_, nullptr,
           nullptr, 0, -2.f, 0, dist, nullptr, KCB_, B_, KCB_, L_);
      argmin_kernel<<<B_, 256, 0, stream>>>(dist, idxb, idxO, q);
      rvq_update_kernel<<<B_, 256, 0, stream>>>(cbF + (size_t)q * KCB_ * L_, idxb, qz32, r32, r16, commitA);
    }
    cvt(qz32, qz16, (size_t)B_ * L_);
  }

  // ---- memory: slots 0 and 1 ----
  gemm(squant16, 256, w16_memproj, 256, F32(IN_MEMPROJ_B), F32(IN_MEMTYPE) + 0,
       nullptr, 0, 1.f, 0, mem32 + 0, nullptr, 768, B_, 256, 256);
  gemm(equant16, 256, w16_memproj, 256, F32(IN_MEMPROJ_B), F32(IN_MEMTYPE) + 256,
       nullptr, 0, 1.f, 0, mem32 + 256, nullptr, 768, B_, 256, 256);
  cvt(mem32, mem16, (size_t)B_ * 3 * D_);

  // ---- decoder query from positions ----
  sinpos_kernel<<<MF_, 128, 0, stream>>>(lengths, h16A);
  gemm(h16A, 256, w16_a1, 256, F32(IN_A1B), nullptr, nullptr, 0, 1.f, 1,
       nullptr, h16B, 256, MF_, 256, 256);
  gemm(h16B, 256, w16_a2, 256, F32(IN_A2B), nullptr, nullptr, 0, 1.f, 0,
       decx32, nullptr, 256, MF_, 256, 256);

  // ---- decoder layers ----
  for (int l = 0; l < 2; ++l) {
    int L = l ? LAY1 : LAY0;
    addln_kernel<<<MF_, 256, 1024, stream>>>(decx32, nullptr, F32(L + 20), F32(L + 21), h16A, nullptr);
    sa_kernel<<<dim3(B_, 4), 128, 0, stream>>>(h16A, saQ16[l], F32(L + 1), saK16[l], F32(L + 3),
                                               saV16[l], F32(L + 5), h16B);
    gemm(h16B, 256, saO16[l], 256, F32(L + 7), nullptr, decx32, 256, 1.f, 0,
         decx32, nullptr, 256, MF_, 256, 256);
    addln_kernel<<<MF_, 256, 1024, stream>>>(decx32, nullptr, F32(L + 22), F32(L + 23), h16A, nullptr);
    gemm(mem16, 256, caK16[l], 256, F32(L + 11), nullptr, nullptr, 0, 1.f, 0,
         nullptr, memK16, 256, B_ * 3, 256, 256);
    gemm(mem16, 256, caV16[l], 256, F32(L + 13), nullptr, nullptr, 0, 1.f, 0,
         nullptr, memV16, 256, B_ * 3, 256, 256);
    ca_kernel<<<dim3(B_, 4), 128, 0, stream>>>(h16A, caQ16[l], F32(L + 9), memK16, memV16, h16B);
    gemm(h16B, 256, caO16[l], 256, F32(L + 15), nullptr, decx32, 256, 1.f, 0,
         decx32, nullptr, 256, MF_, 256, 256);
    addln_kernel<<<MF_, 256, 1024, stream>>>(decx32, nullptr, F32(L + 24), F32(L + 25), h16A, nullptr);
    ffn_kernel<<<MF_ / 32, 256, 0, stream>>>(h16A, f1_16[l], F32(L + 17), f2_16[l], F32(L + 19), decx32);
  }

  // ---- output projection + commit ----
  outproj_kernel<<<(MF_ * 14 + 255) / 256, 256, 0, stream>>>(decx32, F32(IN_OUTW), F32(IN_OUTB), reconO);
  commit_kernel<<<1, 1, 0, stream>>>(commitA, commitO);
}
